// Attention_68719477187
// MI455X (gfx1250) — matrix operand order repaired and checked by reading, not compiled, not run on hardware
//
#include <hip/hip_runtime.h>
#include <hip/hip_bf16.h>

// ---------------------------------------------------------------------------
// MI455X (gfx1250) attention block: QKV GEMM + RoPE + causal flash attention
// (GQA g=4) + output GEMM. Matmuls via v_wmma_f32_16x16x32_bf16 (wave32),
// tile movement via the Tensor Data Mover (tensor_load_to_lds + TENSORcnt),
// double-buffered LDS so TDM DMA overlaps WMMA issue.
// ---------------------------------------------------------------------------

typedef __bf16 bf16;
typedef __attribute__((ext_vector_type(16))) __bf16 v16bf;
typedef __attribute__((ext_vector_type(8)))  __bf16 v8bf;
typedef __attribute__((ext_vector_type(8)))  float  v8f;
typedef __attribute__((ext_vector_type(4)))  unsigned int v4u;
typedef __attribute__((ext_vector_type(8)))  int v8i;
typedef __attribute__((ext_vector_type(4)))  int v4i;

#define WMMA_BF16(a, b, c) \
  __builtin_amdgcn_wmma_f32_16x16x32_bf16(false, (a), false, (b), (short)0, (c), false, false)

#if __has_builtin(__builtin_amdgcn_tensor_load_to_lds) && \
    __has_builtin(__builtin_amdgcn_s_wait_tensorcnt)
#define USE_TDM 1
#else
#define USE_TDM 0
#endif

__device__ __forceinline__ unsigned short f32_to_bf16_bits(float f) {
  unsigned int u = __float_as_uint(f);
  u += 0x7FFFu + ((u >> 16) & 1u);           // round-to-nearest-even
  return (unsigned short)(u >> 16);
}
__device__ __forceinline__ bf16 f32_to_bf16v(float f) {
  unsigned short b = f32_to_bf16_bits(f);
  return __builtin_bit_cast(bf16, b);
}

__device__ __forceinline__ unsigned lds_off(const void* p) {
  return (unsigned)(uintptr_t)p;             // LDS offset lives in addr[31:0]
}

#if USE_TDM
// 2-D tile DMA: global (row stride `stride0` elems, elem = 2B) -> LDS, with
// "pad_amount+1 DWORDs every 2^pad_interval qwords" LDS padding (bank-conflict
// avoidance matches the padded __shared__ tiles below). Issued once per wave.
__device__ __forceinline__ void tdm_load_2d(unsigned lds_addr, const void* gptr,
                                            unsigned tile_d0, unsigned tile_d1,
                                            unsigned tensor_d0, unsigned tensor_d1,
                                            unsigned stride0,
                                            unsigned pad_interval, unsigned pad_amount) {
  unsigned long long ga = (unsigned long long)(uintptr_t)gptr;
  v4u g0;
  g0.x = 1u;                                               // count=1, user D#
  g0.y = lds_addr;                                         // LDS byte address
  g0.z = (unsigned)(ga & 0xFFFFFFFFull);                   // global_addr lo
  g0.w = (unsigned)((ga >> 32) & 0x01FFFFFFull) | (2u << 30);  // hi | type=2
  v8i g1;
  g1[0] = (int)((1u << 16) |                               // data_size = 2B
                (1u << 20) |                               // pad_enable
                (pad_interval << 22) | (pad_amount << 25));
  g1[1] = (int)((tensor_d0 & 0xFFFFu) << 16);              // dim0[15:0] @ b63:48
  g1[2] = (int)((tensor_d0 >> 16) | ((tensor_d1 & 0xFFFFu) << 16));
  g1[3] = (int)((tensor_d1 >> 16) | (tile_d0 << 16));
  g1[4] = (int)(tile_d1 & 0xFFFFu);                        // tile_dim1, tile_dim2=0
  g1[5] = (int)stride0;                                    // dim0 stride lo
  g1[6] = 0;
  g1[7] = 0;                                               // 2-D tensor
  v4i z4 = {};
#if __clang_major__ >= 23
  v8i z8 = {};
  __builtin_amdgcn_tensor_load_to_lds(g0, g1, z4, z4, z8, 0);
#else
  __builtin_amdgcn_tensor_load_to_lds(g0, g1, z4, z4, 0);
#endif
}
#endif

// A-fragment (16xK slice, 16-bit): lane l holds row m=l&15.
// lanes 0-15: K = {kb..kb+7, kb+16..kb+23}; lanes 16-31: +8. (ISA 7.12.2)
__device__ __forceinline__ v16bf load_frag_a(const bf16* base, int ld, int kb, int lane) {
  const bf16* p = base + (lane & 15) * ld + kb + ((lane >> 4) << 3);
  union { v16bf v; v8bf h[2]; } u;
  u.h[0] = *(const v8bf*)(p);
  u.h[1] = *(const v8bf*)(p + 16);
  return u.v;
}
// B-fragment from Bt (N x K layout): lane l holds col n=l&15;
// lanes 0-15: K = {kb..kb+7, kb+16..kb+23}; lanes 16-31: +8.
__device__ __forceinline__ v16bf load_frag_b(const bf16* base, int ld, int kb, int lane) {
  const bf16* p = base + (lane & 15) * ld + kb + ((lane >> 4) << 3);
  union { v16bf v; v8bf h[2]; } u;
  u.h[0] = *(const v8bf*)(p);
  u.h[1] = *(const v8bf*)(p + 16);
  return u.v;
}

// ---------------------------------------------------------------------------
// Kernel 1: f32 -> bf16 conversion (grid-stride)
// ---------------------------------------------------------------------------
__global__ void cvt_f32_bf16(const float* __restrict__ in, unsigned short* __restrict__ out, int n) {
  int i = blockIdx.x * blockDim.x + threadIdx.x;
  int stride = gridDim.x * blockDim.x;
  for (; i < n; i += stride) out[i] = f32_to_bf16_bits(in[i]);
}

// ---------------------------------------------------------------------------
// Kernel 2/5: C(f32, MxN) = A(bf16, MxK) @ Bt(bf16, NxK)^T
// Block tile 128x128, BK=32, 256 threads = 8 waves (4M x 2N), each wave does a
// 32x64 sub-tile: 8 WMMAs per K-step from 6 fragment loads. TDM double-buffer.
// ---------------------------------------------------------------------------
__global__ __launch_bounds__(256) void gemm_bf16tn_f32(
    const bf16* __restrict__ A, const bf16* __restrict__ Bt, float* __restrict__ C,
    int M, int N, int K) {
  __shared__ bf16 As[2][128][40];   // 32 + 8 pad per row (row = 64B + 16B pad)
  __shared__ bf16 Bs[2][128][40];
  const int t = threadIdx.x, lane = t & 31, wave = t >> 5;
  const int mwave = wave >> 1, nwave = wave & 1;  // 4 x 2 wave grid
  const int m0 = blockIdx.y * 128, n0 = blockIdx.x * 128;
  v8f acc0[4] = {}, acc1[4] = {};

#if USE_TDM
  if (wave == 0) {   // one wave drives the DMA engine (EXEC-independent op)
    tdm_load_2d(lds_off(&As[0][0][0]), &A[(size_t)m0 * K], 32, 128,
                (unsigned)K, (unsigned)M, (unsigned)K, 3, 3);
    tdm_load_2d(lds_off(&Bs[0][0][0]), &Bt[(size_t)n0 * K], 32, 128,
                (unsigned)K, (unsigned)N, (unsigned)K, 3, 3);
  }
#endif

  for (int kb = 0; kb < K; kb += 32) {
    const int cur = (kb >> 5) & 1;
#if USE_TDM
    if (wave == 0) {
      if (kb + 32 < K) {
        tdm_load_2d(lds_off(&As[cur ^ 1][0][0]), &A[(size_t)m0 * K + kb + 32], 32, 128,
                    (unsigned)K, (unsigned)M, (unsigned)K, 3, 3);
        tdm_load_2d(lds_off(&Bs[cur ^ 1][0][0]), &Bt[(size_t)n0 * K + kb + 32], 32, 128,
                    (unsigned)K, (unsigned)N, (unsigned)K, 3, 3);
        __builtin_amdgcn_s_wait_tensorcnt(2);   // current tiles landed
      } else {
        __builtin_amdgcn_s_wait_tensorcnt(0);
      }
    }
#else
    {
      const int ar = t >> 1, ac = (t & 1) << 4;
      *(v8bf*)&As[cur][ar][ac]     = *(const v8bf*)&A[(size_t)(m0 + ar) * K + kb + ac];
      *(v8bf*)&As[cur][ar][ac + 8] = *(const v8bf*)&A[(size_t)(m0 + ar) * K + kb + ac + 8];
      *(v8bf*)&Bs[cur][ar][ac]     = *(const v8bf*)&Bt[(size_t)(n0 + ar) * K + kb + ac];
      *(v8bf*)&Bs[cur][ar][ac + 8] = *(const v8bf*)&Bt[(size_t)(n0 + ar) * K + kb + ac + 8];
    }
#endif
    __syncthreads();
    v16bf a0 = load_frag_a(&As[cur][mwave * 32][0],      40, 0, lane);
    v16bf a1 = load_frag_a(&As[cur][mwave * 32 + 16][0], 40, 0, lane);
#pragma unroll
    for (int nf = 0; nf < 4; ++nf) {
      v16bf bfg = load_frag_b(&Bs[cur][nwave * 64 + nf * 16][0], 40, 0, lane);
      acc0[nf] = WMMA_BF16(a0, bfg, acc0[nf]);
      acc1[nf] = WMMA_BF16(a1, bfg, acc1[nf]);
    }
    __syncthreads();   // done reading before next prefetch overwrites buf^1
  }

  // C fragment: VGPR r, lane l -> (m = r + 8*(l>=16), n = l&15)
  const int hf = lane >> 4, nn = lane & 15;
#pragma unroll
  for (int nf = 0; nf < 4; ++nf)
#pragma unroll
    for (int r = 0; r < 8; ++r) {
      size_t col = n0 + nwave * 64 + nf * 16 + nn;
      C[(size_t)(m0 + mwave * 32 + r + 8 * hf) * N + col]      = acc0[nf][r];
      C[(size_t)(m0 + mwave * 32 + 16 + r + 8 * hf) * N + col] = acc1[nf][r];
    }
}

// ---------------------------------------------------------------------------
// Kernel 3: RoPE + repack qkv(f32,[B*S,3072]) ->
//   Q bf16 [b][h][s][64], K bf16 [b][kvh][s][64], Vt bf16 [b][kvh][64][s]
// slots: 0..31 = q heads, 32..39 = k heads, 40..47 = v heads; pair = d/2.
// ---------------------------------------------------------------------------
__global__ void rope_repack(const float* __restrict__ qkv, const float* __restrict__ freqs,
                            unsigned short* __restrict__ Qo, unsigned short* __restrict__ Ko,
                            unsigned short* __restrict__ Vto) {
  int idx = blockIdx.x * 256 + threadIdx.x;      // total = 2*2048*48*32
  int pair = idx & 31;
  int tmp = idx >> 5;
  int slot = tmp % 48; tmp /= 48;
  int s = tmp & 2047;
  int b = tmp >> 11;
  const float* row = qkv + ((size_t)(b * 2048 + s)) * 3072 + slot * 64;
  float v0 = row[2 * pair], v1 = row[2 * pair + 1];
  float o0 = v0, o1 = v1;
  if (slot < 40) {                                // rope applies to q and k
    float c  = freqs[(s * 32 + pair) * 2 + 0];
    float sn = freqs[(s * 32 + pair) * 2 + 1];
    o0 = v0 * c - v1 * sn;
    o1 = v1 * c + v0 * sn;
  }
  if (slot < 32) {
    size_t base = ((size_t)(b * 32 + slot) * 2048 + s) * 64;
    Qo[base + 2 * pair]     = f32_to_bf16_bits(o0);
    Qo[base + 2 * pair + 1] = f32_to_bf16_bits(o1);
  } else if (slot < 40) {
    size_t base = ((size_t)(b * 8 + (slot - 32)) * 2048 + s) * 64;
    Ko[base + 2 * pair]     = f32_to_bf16_bits(o0);
    Ko[base + 2 * pair + 1] = f32_to_bf16_bits(o1);
  } else {
    size_t base = ((size_t)(b * 8 + (slot - 40)) * 64) * 2048;
    Vto[base + (size_t)(2 * pair) * 2048 + s]     = f32_to_bf16_bits(o0);
    Vto[base + (size_t)(2 * pair + 1) * 2048 + s] = f32_to_bf16_bits(o1);
  }
}

// ---------------------------------------------------------------------------
// Kernel 4: causal flash attention, one (b, h, 64 q rows) per block.
// 128 threads = 4 waves; wave w owns q rows q0+16w..+15; kv tiles of 32,
// K/V^T tiles DMA'd by the TDM with double buffering.
// ---------------------------------------------------------------------------
__global__ __launch_bounds__(128) void flash_attn(
    const bf16* __restrict__ Q, const bf16* __restrict__ Kc,
    const bf16* __restrict__ Vt, unsigned short* __restrict__ Oo) {
  __shared__ bf16 Ks[2][32][72];    // K tile: 32 kv rows x 64 d (+8 pad)
  __shared__ bf16 Vs[2][64][40];    // Vt tile: 64 d rows x 32 kv (+8 pad)
  __shared__ bf16 Ps[4][16][40];    // per-wave P round-trip buffer

  const int t = threadIdx.x, lane = t & 31, wave = t >> 5;
  const int qblk = blockIdx.x, h = blockIdx.y, b = blockIdx.z;
  const int kvh = h >> 2;           // GQA: 4 q heads per kv head
  const int q0 = qblk * 64;
  const int hf = lane >> 4, nn = lane & 15;

  const bf16* qptr = Q + ((size_t)(b * 32 + h) * 2048 + q0 + wave * 16) * 64;
  v16bf qf0 = load_frag_a(qptr, 64, 0, lane);     // d = 0..31
  v16bf qf1 = load_frag_a(qptr, 64, 32, lane);    // d = 32..63
  const bf16* kbase = Kc + ((size_t)(b * 8 + kvh) * 2048) * 64;
  const bf16* vbase = Vt + ((size_t)(b * 8 + kvh) * 64) * 2048;

  v8f o[4] = {};
  float mrow[8], lrow[8];
#pragma unroll
  for (int r = 0; r < 8; ++r) { mrow[r] = -3.0e38f; lrow[r] = 0.0f; }

  const int ntiles = (q0 + 64) >> 5;              // uniform across the block

#if USE_TDM
  if (wave == 0) {
    tdm_load_2d(lds_off(&Ks[0][0][0]), kbase, 64, 32, 64, 2048, 64, 4, 3);
    tdm_load_2d(lds_off(&Vs[0][0][0]), vbase, 32, 64, 2048, 64, 2048, 3, 3);
  }
#else
  const int kr = t >> 2, kc = (t & 3) << 4;       // K tile loader mapping
  const int vr = t >> 1, vc = (t & 1) << 4;       // V tile loader mapping
#endif

  for (int jt = 0; jt < ntiles; ++jt) {
    const int j0 = jt << 5;
    const int cur = jt & 1;
#if USE_TDM
    if (wave == 0) {
      if (jt + 1 < ntiles) {
        tdm_load_2d(lds_off(&Ks[cur ^ 1][0][0]), &kbase[(size_t)(j0 + 32) * 64],
                    64, 32, 64, 2048, 64, 4, 3);
        tdm_load_2d(lds_off(&Vs[cur ^ 1][0][0]), &vbase[j0 + 32],
                    32, 64, 2048, 64, 2048, 3, 3);
        __builtin_amdgcn_s_wait_tensorcnt(2);     // current tiles landed
      } else {
        __builtin_amdgcn_s_wait_tensorcnt(0);
      }
    }
#else
    *(v8bf*)&Ks[cur][kr][kc]     = *(const v8bf*)&kbase[(size_t)(j0 + kr) * 64 + kc];
    *(v8bf*)&Ks[cur][kr][kc + 8] = *(const v8bf*)&kbase[(size_t)(j0 + kr) * 64 + kc + 8];
    *(v8bf*)&Vs[cur][vr][vc]     = *(const v8bf*)&vbase[(size_t)vr * 2048 + j0 + vc];
    *(v8bf*)&Vs[cur][vr][vc + 8] = *(const v8bf*)&vbase[(size_t)vr * 2048 + j0 + vc + 8];
#endif
    __syncthreads();

    // S = Q @ K^T  (16 q rows x 32 kv cols), f32 accumulate
    v8f s0 = {}, s1 = {};
    s0 = WMMA_BF16(qf0, load_frag_b(&Ks[cur][0][0],  72, 0,  lane), s0);
    s0 = WMMA_BF16(qf1, load_frag_b(&Ks[cur][0][0],  72, 32, lane), s0);
    s1 = WMMA_BF16(qf0, load_frag_b(&Ks[cur][16][0], 72, 0,  lane), s1);
    s1 = WMMA_BF16(qf1, load_frag_b(&Ks[cur][16][0], 72, 32, lane), s1);

    // scale + causal mask
    const int mg0 = q0 + wave * 16 + 8 * hf;
#pragma unroll
    for (int r = 0; r < 8; ++r) {
      const int mg = mg0 + r;
      float a0 = s0[r] * 0.125f;                  // HEAD_DIM^-0.5
      float a1 = s1[r] * 0.125f;
      if (j0 + nn > mg)      a0 = -3.0e38f;
      if (j0 + 16 + nn > mg) a1 = -3.0e38f;
      s0[r] = a0; s1[r] = a1;
    }

    // online softmax: rows live across 16-lane halves for each acc VGPR
#pragma unroll
    for (int r = 0; r < 8; ++r) {
      float rm = fmaxf(s0[r], s1[r]);
#pragma unroll
      for (int off = 8; off >= 1; off >>= 1)
        rm = fmaxf(rm, __shfl_xor(rm, off, 32));
      float mnew = fmaxf(mrow[r], rm);
      float corr = __expf(mrow[r] - mnew);
      float p0 = __expf(s0[r] - mnew);
      float p1 = __expf(s1[r] - mnew);
      float ps = p0 + p1;
#pragma unroll
      for (int off = 8; off >= 1; off >>= 1)
        ps += __shfl_xor(ps, off, 32);
      lrow[r] = lrow[r] * corr + ps;
      mrow[r] = mnew;
#pragma unroll
      for (int nf = 0; nf < 4; ++nf) o[nf][r] *= corr;
      s0[r] = p0; s1[r] = p1;
    }

    // C-layout -> A-layout via per-wave LDS buffer
#pragma unroll
    for (int r = 0; r < 8; ++r) {
      Ps[wave][r + 8 * hf][nn]      = f32_to_bf16v(s0[r]);
      Ps[wave][r + 8 * hf][16 + nn] = f32_to_bf16v(s1[r]);
    }
    asm volatile("s_wait_dscnt 0" ::: "memory");   // same-wave LDS RAW

    // O += P(16x32) @ V(32x64); Vt tile gives contiguous-K B fragments
    v16bf pf = load_frag_a(&Ps[wave][0][0], 40, 0, lane);
#pragma unroll
    for (int nf = 0; nf < 4; ++nf)
      o[nf] = WMMA_BF16(pf, load_frag_b(&Vs[cur][nf * 16][0], 40, 0, lane), o[nf]);

    __syncthreads();   // done reading before next prefetch overwrites buf^1
  }

  // epilogue: O /= l, write bf16 attn[b*S+s][h*64+d] for the final GEMM
#pragma unroll
  for (int r = 0; r < 8; ++r) {
    float inv = 1.0f / lrow[r];
    size_t row = (size_t)b * 2048 + q0 + wave * 16 + r + 8 * hf;
#pragma unroll
    for (int nf = 0; nf < 4; ++nf)
      Oo[row * 2048 + h * 64 + nf * 16 + nn] = f32_to_bf16_bits(o[nf][r] * inv);
  }
}

// ---------------------------------------------------------------------------
extern "C" void kernel_launch(void* const* d_in, const int* in_sizes, int n_in,
                              void* d_out, int out_size, void* d_ws, size_t ws_size,
                              hipStream_t stream) {
  const float* x     = (const float*)d_in[0];   // (2, 2048, 2048)
  const float* freqs = (const float*)d_in[1];   // (2048, 32, 2)
  const float* wqkv  = (const float*)d_in[2];   // (3072, 2048)
  const float* wo    = (const float*)d_in[3];   // (2048, 2048)

  char* ws = (char*)d_ws;
  size_t off = 0;
  auto alloc = [&](size_t bytes) -> void* {
    void* p = ws + off;
    off += (bytes + 255) & ~(size_t)255;
    return p;
  };
  unsigned short* x_bf    = (unsigned short*)alloc(4096ull * 2048 * 2);
  unsigned short* wqkv_bf = (unsigned short*)alloc(3072ull * 2048 * 2);
  unsigned short* wo_bf   = (unsigned short*)alloc(2048ull * 2048 * 2);
  float*          qkv     = (float*)         alloc(4096ull * 3072 * 4);
  unsigned short* Qb      = (unsigned short*)alloc(2ull * 32 * 2048 * 64 * 2);
  unsigned short* Kb      = (unsigned short*)alloc(2ull * 8 * 2048 * 64 * 2);
  unsigned short* Vtb     = (unsigned short*)alloc(2ull * 8 * 64 * 2048 * 2);
  unsigned short* attn    = (unsigned short*)alloc(4096ull * 2048 * 2);

  cvt_f32_bf16<<<4096, 256, 0, stream>>>(x,    x_bf,    4096 * 2048);
  cvt_f32_bf16<<<4096, 256, 0, stream>>>(wqkv, wqkv_bf, 3072 * 2048);
  cvt_f32_bf16<<<4096, 256, 0, stream>>>(wo,   wo_bf,   2048 * 2048);

  // qkv = x @ wqkv^T   (M=4096, N=3072, K=2048), f32 accum out for RoPE
  gemm_bf16tn_f32<<<dim3(3072 / 128, 4096 / 128), 256, 0, stream>>>(
      (const bf16*)x_bf, (const bf16*)wqkv_bf, qkv, 4096, 3072, 2048);

  rope_repack<<<24576, 256, 0, stream>>>(qkv, freqs, Qb, Kb, Vtb);

  flash_attn<<<dim3(32, 32, 2), 128, 0, stream>>>(
      (const bf16*)Qb, (const bf16*)Kb, (const bf16*)Vtb, attn);

  // out = attn @ wo^T  (M=4096, N=2048, K=2048) -> f32 d_out
  gemm_bf16tn_f32<<<dim3(2048 / 128, 4096 / 128), 256, 0, stream>>>(
      (const bf16*)attn, (const bf16*)wo_bf, (float*)d_out, 4096, 2048, 2048);
}
